// PointnetSAModule_32581621908084
// MI455X (gfx1250) — matrix operand order repaired and checked by reading, not compiled, not run on hardware
//
#include <hip/hip_runtime.h>

typedef __attribute__((ext_vector_type(16))) __bf16 v16bf;
typedef __attribute__((ext_vector_type(8)))  float  v8f;

__device__ __forceinline__ unsigned short f2bf_u(float f) {
    unsigned x = __float_as_uint(f);
    unsigned r = x + 0x7fffu + ((x >> 16) & 1u);   // round-to-nearest-even
    return (unsigned short)(r >> 16);
}
__device__ __forceinline__ float bf2f_u(unsigned short u) {
    return __uint_as_float(((unsigned)u) << 16);
}

// Tiled (WMMA-B native) offset for element (channel c, column n); KG = CinPad/32.
__device__ __forceinline__ size_t btile_off(int c, int n, int KG) {
    int kg = c >> 5, w = c & 31;
    return (((size_t)(n >> 4) * KG + kg) * 32 + ((n & 15) + ((w >> 4) << 4))) * 16 + (w & 15);
}

// ---------------------------------------------------------------- pack fp32 W[Cout][Cin] -> A-tiled bf16
__global__ void k_wpack(const float* __restrict__ src, unsigned short* __restrict__ dst,
                        int Cin, int Cout, int CinPad) {
    int i = blockIdx.x * 256 + threadIdx.x;
    if (i >= Cout * CinPad) return;
    int e     = i & 15;
    int lane  = (i >> 4) & 31;
    int kgAll = i >> 9;
    int KG    = CinPad >> 5;
    int kg    = kgAll % KG;
    int mt    = kgAll / KG;
    int r = e >> 1, h = e & 1;
    int k = kg * 32 + e + ((lane >= 16) ? 16 : 0);
    int m = mt * 16 + (lane & 15);
    float v = (k < Cin) ? src[(size_t)m * Cin + k] : 0.0f;
    dst[i] = f2bf_u(v);
}

// ---------------------------------------------------------------- zero floats
__global__ void k_zero(float* __restrict__ p, int n) {
    int i = blockIdx.x * 256 + threadIdx.x;
    if (i < n) p[i] = 0.0f;
}

// ---------------------------------------------------------------- per-batch mean of xyz
__global__ void k_mean(const float* __restrict__ xyz, float* __restrict__ meanb) {
    int b = blockIdx.x, t = threadIdx.x;
    const float* xb = xyz + (size_t)b * 8192 * 3;
    float sx = 0.f, sy = 0.f, sz = 0.f;
    for (int j = t; j < 8192; j += 256) {
        sx += xb[j * 3 + 0]; sy += xb[j * 3 + 1]; sz += xb[j * 3 + 2];
    }
    __shared__ float rx[256], ry[256], rz[256];
    rx[t] = sx; ry[t] = sy; rz[t] = sz;
    __syncthreads();
    for (int off = 128; off >= 1; off >>= 1) {
        if (t < off) { rx[t] += rx[t + off]; ry[t] += ry[t + off]; rz[t] += rz[t + off]; }
        __syncthreads();
    }
    if (t == 0) {
        meanb[b * 3 + 0] = rx[0] / 8192.f;
        meanb[b * 3 + 1] = ry[0] / 8192.f;
        meanb[b * 3 + 2] = rz[0] / 8192.f;
    }
}

// ---------------------------------------------------------------- farthest point sampling
__global__ void __launch_bounds__(1024) k_fps(const float* __restrict__ xyz,
                                              const float* __restrict__ meanb,
                                              float* __restrict__ out) {
    int b = blockIdx.x;
    int t = threadIdx.x;
    const float* xb = xyz + (size_t)b * 8192 * 3;

    float px[8], py[8], pz[8], dist[8];
#pragma unroll
    for (int i = 0; i < 8; ++i) {
        int j = i * 1024 + t;
        float x, y, z;
        if (j == 0) { x = meanb[b*3]; y = meanb[b*3+1]; z = meanb[b*3+2]; }
        else        { x = xb[(j-1)*3]; y = xb[(j-1)*3+1]; z = xb[(j-1)*3+2]; }
        px[i] = x; py[i] = y; pz[i] = z; dist[i] = 1e10f;
    }
    float ex = 0.f, ey = 0.f, ez = 0.f, edist = 1e10f;  // extra point j=8192 (owner t==0)
    if (t == 0) { ex = xb[8191*3]; ey = xb[8191*3+1]; ez = xb[8191*3+2]; }

    __shared__ float cb[3];
    __shared__ float rv[32];
    __shared__ int   ri[32];
    __shared__ int   sfar;

    int far = 0;
    for (int step = 0; step < 1024; ++step) {
        if (t == 0) {
            float cx, cy, cz;
            if (far == 0) { cx = meanb[b*3]; cy = meanb[b*3+1]; cz = meanb[b*3+2]; }
            else          { cx = xb[(far-1)*3]; cy = xb[(far-1)*3+1]; cz = xb[(far-1)*3+2]; }
            cb[0] = cx; cb[1] = cy; cb[2] = cz;
            float* o = out + ((size_t)b * 1024 + step) * 3;
            o[0] = cx; o[1] = cy; o[2] = cz;
        }
        __syncthreads();
        float cx = cb[0], cy = cb[1], cz = cb[2];

        float best = -1.0f; int bidx = 0x7fffffff;
#pragma unroll
        for (int i = 0; i < 8; ++i) {
            float dx = px[i]-cx, dy = py[i]-cy, dz = pz[i]-cz;
            float d2 = dx*dx + dy*dy + dz*dz;
            dist[i] = fminf(dist[i], d2);
            if (dist[i] > best) { best = dist[i]; bidx = i * 1024 + t; }
        }
        if (t == 0) {
            float dx = ex-cx, dy = ey-cy, dz = ez-cz;
            float d2 = dx*dx + dy*dy + dz*dz;
            edist = fminf(edist, d2);
            if (edist > best) { best = edist; bidx = 8192; }
        }
#pragma unroll
        for (int off = 16; off >= 1; off >>= 1) {   // ties -> smaller index (jnp.argmax)
            float ov = __shfl_xor(best, off);
            int   oi = __shfl_xor(bidx, off);
            if (ov > best || (ov == best && oi < bidx)) { best = ov; bidx = oi; }
        }
        int wv = t >> 5;
        if ((t & 31) == 0) { rv[wv] = best; ri[wv] = bidx; }
        __syncthreads();
        if (t < 32) {
            float v2 = rv[t]; int i2 = ri[t];
#pragma unroll
            for (int off = 16; off >= 1; off >>= 1) {
                float ov = __shfl_xor(v2, off);
                int   oi = __shfl_xor(i2, off);
                if (ov > v2 || (ov == v2 && oi < i2)) { v2 = ov; i2 = oi; }
            }
            if (t == 0) sfar = i2;
        }
        __syncthreads();
        far = sfar;
    }
}

// ---------------------------------------------------------------- ball query (1 wave / query)
__global__ void k_ball(const float* __restrict__ xyz, const float* __restrict__ newxyz,
                       int* __restrict__ idxArr, float r2, int nsample) {
    int lane = threadIdx.x & 31;
    int wv   = threadIdx.x >> 5;
    int q    = blockIdx.x * 8 + wv;          // b*1024 + s
    int b    = q >> 10;
    const float* xb = xyz + (size_t)b * 8192 * 3;
    const float* qq = newxyz + (size_t)q * 3;
    float qx = qq[0], qy = qq[1], qz = qq[2];
    int* ob = idxArr + (size_t)q * nsample;

    int cnt = 0;
    int firstJ = 0x7fffffff;
    for (int j0 = 0; j0 < 8192 && cnt < nsample; j0 += 32) {
        int j = j0 + lane;
        float dx = xb[j*3+0]-qx, dy = xb[j*3+1]-qy, dz = xb[j*3+2]-qz;
        float d2 = dx*dx + dy*dy + dz*dz;
        bool in = d2 < r2;
        unsigned mask = (unsigned)__ballot(in);
        int pre = __popc(mask & ((1u << lane) - 1u));
        int pos = cnt + pre;
        if (in && pos < nsample) ob[pos] = j;
        int cand = (in && pos == 0) ? j : 0x7fffffff;
#pragma unroll
        for (int off = 16; off >= 1; off >>= 1)
            cand = min(cand, __shfl_xor(cand, off));
        firstJ = min(firstJ, cand);
        cnt += __popc(mask);
    }
    if (cnt == 0) {
        for (int p = lane; p < nsample; p += 32) ob[p] = 0;
    } else if (cnt < nsample) {
        for (int p = cnt + lane; p < nsample; p += 32) ob[p] = firstJ;
    }
}

// ---------------------------------------------------------------- gather -> X0 (B-tiled bf16, 96 padded rows)
__global__ void k_gather(const float* __restrict__ xyz, const float* __restrict__ feat,
                         const float* __restrict__ newxyz, const int* __restrict__ idxArr,
                         unsigned short* __restrict__ Xt, int kShift, int Nt) {
    int n = blockIdx.x * 256 + threadIdx.x;
    if (n >= Nt) return;
    int bs = n >> kShift;           // b*1024 + s
    int b  = bs >> 10;
    int j  = idxArr[n];
    const float* p = xyz + ((size_t)b * 8192 + j) * 3;
    const float* q = newxyz + (size_t)bs * 3;
    Xt[btile_off(0, n, 3)] = f2bf_u(p[0] - q[0]);
    Xt[btile_off(1, n, 3)] = f2bf_u(p[1] - q[1]);
    Xt[btile_off(2, n, 3)] = f2bf_u(p[2] - q[2]);
    const float* fb = feat + (size_t)b * 64 * 8192 + j;
#pragma unroll 4
    for (int c = 0; c < 64; ++c)
        Xt[btile_off(3 + c, n, 3)] = f2bf_u(fb[(size_t)c * 8192]);
#pragma unroll
    for (int c = 67; c < 96; ++c)   // zero the K-padding rows
        Xt[btile_off(c, n, 3)] = 0;
}

// ---------------------------------------------------------------- GEMM (A,B pre-tiled; no guards)
// Each wave: A fragments register-resident, 8 consecutive N-tiles, one stats
// reduction at the end (register-accumulated Sum/SumSq -> fixed-order butterflies).
// MODE 0: store Y into next layer's B-tiled layout + channel sums
// MODE 1: channel sums only      MODE 2: affine+ReLU+max-over-K via atomicMax
template <int KSTEPS, int MODE>
__global__ void k_gemm(const unsigned short* __restrict__ Xt,
                       const unsigned short* __restrict__ Wt,
                       unsigned short* __restrict__ Yt,
                       float* __restrict__ psum, float* __restrict__ psq,
                       const float* __restrict__ scale, const float* __restrict__ shift,
                       float* __restrict__ outF,
                       int kShift, int cbase, int KGout) {
    const int TPW = 8;                        // n-tiles per wave
    int lane = threadIdx.x & 31;
    int wv   = threadIdx.x >> 5;
    int half = lane >> 4;
    int l15  = lane & 15;
    int mt   = blockIdx.y;
    int M0   = mt * 16;
    int tn0  = (blockIdx.x * 8 + wv) * TPW;

    const v16bf* Af = (const v16bf*)Wt + (size_t)mt * KSTEPS * 32 + lane;
    v16bf a[KSTEPS];
#pragma unroll
    for (int kg = 0; kg < KSTEPS; ++kg) a[kg] = Af[kg * 32];

    float sc_[8], sh_[8];
    if (MODE == 2) {
#pragma unroll
        for (int r = 0; r < 8; ++r) {
            int m = M0 + r + half * 8;
            sc_[r] = scale[m]; sh_[r] = shift[m];
        }
    }

    float sacc[8] = {}, qacc[8] = {};
    for (int t = 0; t < TPW; ++t) {
        int tn = tn0 + t;
        const v16bf* Bf = (const v16bf*)Xt + (size_t)tn * KSTEPS * 32 + lane;
        v8f c = {};
#pragma unroll
        for (int kg = 0; kg < KSTEPS; ++kg)
            c = __builtin_amdgcn_wmma_f32_16x16x32_bf16(false, a[kg], false, Bf[kg * 32],
                                                        (short)0, c, false, false);
        int ncol = tn * 16 + l15;
        if (MODE == 2) {
#pragma unroll
            for (int r = 0; r < 8; ++r) {
                int m = M0 + r + half * 8;
                float v = fmaxf(c[r] * sc_[r] + sh_[r], 0.0f);
                int b_ = ncol >> (10 + kShift);
                int s_ = (ncol >> kShift) & 1023;
                int oi = ((b_ * 384) + cbase + m) * 1024 + s_;
                atomicMax((int*)(outF + oi), __float_as_int(v));   // v >= 0
            }
        } else {
#pragma unroll
            for (int r = 0; r < 8; ++r) {
                float v = c[r];
                if (MODE == 0) {
                    int m = M0 + r + half * 8;
                    size_t o = (((size_t)tn * KGout + (m >> 5)) * 32 +
                                (l15 + (((m >> 4) & 1) << 4))) * 16 + (m & 15);
                    Yt[o] = f2bf_u(v);
                }
                sacc[r] += v;
                qacc[r] += v * v;
            }
        }
    }
    if (MODE == 2) return;

    __shared__ float lsum[8][16];
    __shared__ float lsq[8][16];
#pragma unroll
    for (int r = 0; r < 8; ++r) {
        float v = sacc[r], q = qacc[r];
#pragma unroll
        for (int off = 1; off < 16; off <<= 1) {   // fixed-order butterfly in 16-lane halves
            v += __shfl_xor(v, off, 16);
            q += __shfl_xor(q, off, 16);
        }
        if (l15 == 0) {
            int row = r + half * 8;
            lsum[wv][row] = v;
            lsq [wv][row] = q;
        }
    }
    __syncthreads();
    int tt = threadIdx.x;
    if (tt < 16) {
        float s = 0.f, sq = 0.f;
#pragma unroll
        for (int w = 0; w < 8; ++w) { s += lsum[w][tt]; sq += lsq[w][tt]; }
        size_t slot = ((size_t)blockIdx.y * gridDim.x + blockIdx.x) * 16 + tt;
        psum[slot] = s; psq[slot] = sq;
    }
}

// ---------------------------------------------------------------- fold partials -> scale/shift
__global__ void k_stats(const float* __restrict__ psum, const float* __restrict__ psq,
                        const float* __restrict__ g, const float* __restrict__ bb,
                        float* __restrict__ scale, float* __restrict__ shift,
                        int Cout, int gridNx, float invCnt) {
    int c = threadIdx.x;
    if (c >= Cout) return;
    int mt = c >> 4, row = c & 15;
    float s = 0.f, sq = 0.f;
    for (int x = 0; x < gridNx; ++x) {
        size_t slot = ((size_t)mt * gridNx + x) * 16 + row;
        s += psum[slot]; sq += psq[slot];
    }
    float mean = s * invCnt;
    float var  = sq * invCnt - mean * mean;
    float sc   = g[c] / sqrtf(var + 1e-5f);
    scale[c] = sc;
    shift[c] = bb[c] - mean * sc;
}

// ---------------------------------------------------------------- in-place BN affine + ReLU on tiled bf16
__global__ void k_bnrelu(unsigned short* __restrict__ y, const float* __restrict__ scale,
                         const float* __restrict__ shift, int KG, long total) {
    long i = (long)blockIdx.x * 256 + threadIdx.x;
    if (i >= total) return;
    int e    = (int)(i & 15);
    int lane = (int)((i >> 4) & 31);
    int kg   = (int)((i >> 9) % KG);
    int c    = kg * 32 + ((lane >> 4) << 4) + e;
    float v = bf2f_u(y[i]) * scale[c] + shift[c];
    y[i] = f2bf_u(fmaxf(v, 0.0f));
}

// ================================================================ host
extern "C" void kernel_launch(void* const* d_in, const int* in_sizes, int n_in,
                              void* d_out, int out_size, void* d_ws, size_t ws_size,
                              hipStream_t stream) {
    (void)in_sizes; (void)n_in; (void)out_size; (void)ws_size;
    const float* xyz  = (const float*)d_in[0];
    const float* feat = (const float*)d_in[1];
    float* out = (float*)d_out;
    char*  ws  = (char*)d_ws;

    const int B = 8, S = 1024;
    size_t off = 0;
    auto alloc = [&](size_t bytes) { size_t r = off; off += (bytes + 255) & ~(size_t)255; return r; };

    size_t oMean  = alloc((size_t)B * 3 * 4);
    size_t oIdx0  = alloc((size_t)B * S * 32 * 4);
    size_t oIdx1  = alloc((size_t)B * S * 64 * 4);
    size_t oWbf   = alloc((size_t)65536 * 2);
    size_t oScale = alloc(256 * 4);
    size_t oShift = alloc(256 * 4);
    size_t oPsum  = alloc((size_t)16 * 512 * 16 * 4);
    size_t oPsq   = alloc((size_t)16 * 512 * 16 * 4);
    size_t Nt1    = (size_t)B * S * 64;
    size_t oR0    = alloc(96  * Nt1 * 2);   // CinPad = 96 for layer 0
    size_t oR1    = alloc(64  * Nt1 * 2);
    size_t oR2    = alloc(128 * Nt1 * 2);

    float* mean          = (float*)(ws + oMean);
    unsigned short* wbf  = (unsigned short*)(ws + oWbf);
    float* scaleA        = (float*)(ws + oScale);
    float* shiftA        = (float*)(ws + oShift);
    float* psum          = (float*)(ws + oPsum);
    float* psq           = (float*)(ws + oPsq);
    unsigned short* R[3] = { (unsigned short*)(ws + oR0),
                             (unsigned short*)(ws + oR1),
                             (unsigned short*)(ws + oR2) };

    // weight packing: fp32 [Cout][Cin] -> A-tiled bf16 [Cout/16][CinPad/32][32][16]
    const int wIn   [6] = {2, 5, 8, 11, 14, 17};
    const int wCin  [6] = {67, 64, 64, 67, 64, 128};
    const int wCout [6] = {64, 64, 128, 64, 128, 256};
    const int wCpad [6] = {96, 64, 64, 96, 64, 128};
    const int wOff  [6] = {0, 6144, 10240, 18432, 24576, 32768};
    for (int i = 0; i < 6; ++i) {
        int n = wCout[i] * wCpad[i];
        k_wpack<<<(n + 255) / 256, 256, 0, stream>>>((const float*)d_in[wIn[i]],
                                                     wbf + wOff[i], wCin[i], wCout[i], wCpad[i]);
    }

    k_mean<<<B, 256, 0, stream>>>(xyz, mean);
    k_fps<<<B, 1024, 0, stream>>>(xyz, mean, out);
    k_zero<<<(B * 384 * 1024 + 255) / 256, 256, 0, stream>>>(out + (size_t)B * S * 3, B * 384 * 1024);

    for (int bi = 0; bi < 2; ++bi) {
        const int   K      = bi ? 64 : 32;
        const int   kShift = bi ? 6 : 5;
        const float radius = bi ? 0.8f : 0.4f;
        int* idxArr = (int*)(ws + (bi ? oIdx1 : oIdx0));
        const int Nt = B * S * K;
        int cout[3], ksteps[3];
        if (bi == 0) { cout[0]=64;  cout[1]=64;  cout[2]=128; ksteps[0]=3; ksteps[1]=2; ksteps[2]=2; }
        else         { cout[0]=64;  cout[1]=128; cout[2]=256; ksteps[0]=3; ksteps[1]=2; ksteps[2]=4; }
        const int cbase = bi ? 128 : 0;

        k_ball<<<(B * S) / 8, 256, 0, stream>>>(xyz, out, idxArr, radius * radius, K);
        k_gather<<<(Nt + 255) / 256, 256, 0, stream>>>(xyz, feat, out, idxArr, R[0], kShift, Nt);

        for (int L = 0; L < 3; ++L) {
            unsigned short* Wp = wbf + wOff[bi * 3 + L];
            const float* gp = (const float*)d_in[2 + bi * 9 + L * 3 + 1];
            const float* bp = (const float*)d_in[2 + bi * 9 + L * 3 + 2];
            const int gridNx = (Nt / 16) / 64;          // 8 waves x 8 tiles per block
            dim3 grid(gridNx, cout[L] / 16);
            const float invCnt = 1.0f / (float)Nt;
            const int KGout = cout[L] / 32;
            float* outFeat = out + (size_t)B * S * 3;

            auto launch = [&](int mode, unsigned short* Yp) {
                int key = ksteps[L] * 10 + mode;
                switch (key) {
                case 20: k_gemm<2,0><<<grid, 256, 0, stream>>>(R[L], Wp, Yp, psum, psq, scaleA, shiftA, outFeat, kShift, cbase, KGout); break;
                case 21: k_gemm<2,1><<<grid, 256, 0, stream>>>(R[L], Wp, Yp, psum, psq, scaleA, shiftA, outFeat, kShift, cbase, KGout); break;
                case 22: k_gemm<2,2><<<grid, 256, 0, stream>>>(R[L], Wp, Yp, psum, psq, scaleA, shiftA, outFeat, kShift, cbase, KGout); break;
                case 30: k_gemm<3,0><<<grid, 256, 0, stream>>>(R[L], Wp, Yp, psum, psq, scaleA, shiftA, outFeat, kShift, cbase, KGout); break;
                case 31: k_gemm<3,1><<<grid, 256, 0, stream>>>(R[L], Wp, Yp, psum, psq, scaleA, shiftA, outFeat, kShift, cbase, KGout); break;
                case 32: k_gemm<3,2><<<grid, 256, 0, stream>>>(R[L], Wp, Yp, psum, psq, scaleA, shiftA, outFeat, kShift, cbase, KGout); break;
                case 40: k_gemm<4,0><<<grid, 256, 0, stream>>>(R[L], Wp, Yp, psum, psq, scaleA, shiftA, outFeat, kShift, cbase, KGout); break;
                case 41: k_gemm<4,1><<<grid, 256, 0, stream>>>(R[L], Wp, Yp, psum, psq, scaleA, shiftA, outFeat, kShift, cbase, KGout); break;
                default: k_gemm<4,2><<<grid, 256, 0, stream>>>(R[L], Wp, Yp, psum, psq, scaleA, shiftA, outFeat, kShift, cbase, KGout); break;
                }
            };

            if (L < 2) {
                launch(0, R[L + 1]);
                k_stats<<<1, 256, 0, stream>>>(psum, psq, gp, bp, scaleA, shiftA,
                                               cout[L], gridNx, invCnt);
                long total = (long)cout[L] * Nt;
                k_bnrelu<<<(unsigned)((total + 255) / 256), 256, 0, stream>>>(R[L + 1], scaleA,
                                                                              shiftA, KGout, total);
            } else {
                launch(1, nullptr);
                k_stats<<<1, 256, 0, stream>>>(psum, psq, gp, bp, scaleA, shiftA,
                                               cout[L], gridNx, invCnt);
                launch(2, nullptr);
            }
        }
    }
}
